// BertForRank_70068096467942
// MI455X (gfx1250) — matrix operand order repaired and checked by reading, not compiled, not run on hardware
//
#include <hip/hip_runtime.h>
#include <hip/hip_bf16.h>

// ---------------------------------------------------------------------------
// BertForRank pipeline for MI455X (gfx1250, wave32, WMMA + async-LDS DMA).
// ---------------------------------------------------------------------------

typedef __attribute__((ext_vector_type(16))) _Float16 v16h;
typedef __attribute__((ext_vector_type(8)))  float    v8f;
typedef __attribute__((ext_vector_type(4)))  float    f4;

#define Bb 8
#define Ss 256
#define Hh 768
#define Dd 128
#define Ll 64
#define BS 2048   // B*S
#define BL 512    // B*L

union VA { v16h v; _Float16 h[16]; };
union VC { v8f  v; float     f[8]; };

// A-matrix (16-bit, 16x32) half-index -> K  (ISA 7.12.2): pairs at
// base = {0,2,4,6,16,18,20,22} + 8*(lane>>4), elements base, base+1.
__device__ __forceinline__ int a_kbase(int j, int g) {
    return ((j < 4) ? (2 * j) : (8 + 2 * j)) + 8 * g;
}
__device__ __forceinline__ int a_kidx(int h, int g) {
    return a_kbase(h >> 1, g) + (h & 1);
}
// 16-bit operand half-index -> K, contiguous order (k = h + 16*g), used for both A and B
__device__ __forceinline__ int b_kidx(int h, int g) { return h + 16 * g; }

__device__ __forceinline__ v8f wmma_f16(v16h a, v16h b, v8f c) {
    return __builtin_amdgcn_wmma_f32_16x16x32_f16(
        /*neg_a=*/false, a, /*neg_b=*/false, b,
        /*c_mod=*/(short)0, c, /*reuse_a=*/false, /*reuse_b=*/false);
}

// ---------------------------------------------------------------------------
// C[m,n] = sum_k A[m,k]*B[k,n] (+bias[n]);  A:[M,K] rm, B:[K,N] rm, batched.
// One wave per 16x16 tile.  B tile staged through LDS (coalesced float4 rows).
// ---------------------------------------------------------------------------
__global__ void wmma_gemm_nn(const float* __restrict__ A,
                             const float* __restrict__ Bm,
                             const float* __restrict__ bias,
                             float* __restrict__ C,
                             int M, int N, int K,
                             long bsA, long bsB, long bsC) {
    __shared__ float4 tB4[32 * 4];           // 32 rows x 16 cols
    float* tB = (float*)tB4;
    long z = blockIdx.z;
    A  += z * bsA;  Bm += z * bsB;  C += z * bsC;
    int tm = blockIdx.x, tn = blockIdx.y;
    int lane = threadIdx.x & 31;
    int m = lane & 15, g = lane >> 4;
    int arow = tm * 16 + m;
    const float* arp = A + (long)arow * K;
    VC acc; acc.v = v8f{0.f,0.f,0.f,0.f,0.f,0.f,0.f,0.f};
    for (int kt = 0; kt < K; kt += 32) {
        // stage B tile: lane r loads row kt+r, cols [tn*16, tn*16+16)
        const float4* src = (const float4*)(Bm + (long)(kt + lane) * N + tn * 16);
#pragma unroll
        for (int q = 0; q < 4; ++q) tB4[lane * 4 + q] = src[q];
        if (kt + 32 < K) __builtin_prefetch(arp + kt + 32, 0, 3);
        __syncthreads();
        VA ua, ub;
#pragma unroll
        for (int j = 0; j < 8; ++j) {
            const float2 p = *(const float2*)(arp + kt + 16 * g + 2 * j);
            ua.h[2 * j]     = (_Float16)p.x;
            ua.h[2 * j + 1] = (_Float16)p.y;
        }
#pragma unroll
        for (int h = 0; h < 16; ++h)
            ub.h[h] = (_Float16)tB[(h + 16 * g) * 16 + m];
        acc.v = wmma_f16(ua.v, ub.v, acc.v);
        __syncthreads();
    }
    int c = tn * 16 + (lane & 15);
    float bv = bias ? bias[c] : 0.f;
    float* cp = C + (long)(tm * 16 + 8 * g) * N + c;
#pragma unroll
    for (int i = 0; i < 8; ++i) cp[(long)i * N] = acc.f[i] + bv;
}

// ---------------------------------------------------------------------------
// C[m,n] = sum_k A[m,k]*B[n,k];  A:[M,K] rm, B:[N,K] rm (both K-contiguous).
// ---------------------------------------------------------------------------
__global__ void wmma_gemm_nt(const float* __restrict__ A,
                             const float* __restrict__ Bm,
                             float* __restrict__ C,
                             int M, int N, int K) {
    int tm = blockIdx.x, tn = blockIdx.y;
    int lane = threadIdx.x & 31;
    int m = lane & 15, g = lane >> 4;
    const float* arp = A + (long)(tm * 16 + m) * K;
    const float* brp = Bm + (long)(tn * 16 + m) * K;
    VC acc; acc.v = v8f{0.f,0.f,0.f,0.f,0.f,0.f,0.f,0.f};
    for (int kt = 0; kt < K; kt += 32) {
        if (kt + 32 < K) {
            __builtin_prefetch(arp + kt + 32, 0, 3);
            __builtin_prefetch(brp + kt + 32, 0, 3);
        }
        VA ua, ub;
#pragma unroll
        for (int j = 0; j < 8; ++j) {
            const float2 p = *(const float2*)(arp + kt + 16 * g + 2 * j);
            ua.h[2 * j]     = (_Float16)p.x;
            ua.h[2 * j + 1] = (_Float16)p.y;
        }
        const float4* bp = (const float4*)(brp + kt + 16 * g);
#pragma unroll
        for (int q = 0; q < 4; ++q) {
            float4 v4 = bp[q];
            ub.h[4 * q + 0] = (_Float16)v4.x;
            ub.h[4 * q + 1] = (_Float16)v4.y;
            ub.h[4 * q + 2] = (_Float16)v4.z;
            ub.h[4 * q + 3] = (_Float16)v4.w;
        }
        acc.v = wmma_f16(ua.v, ub.v, acc.v);
    }
    float* cp = C + (long)(tm * 16 + 8 * g) * N + tn * 16 + (lane & 15);
#pragma unroll
    for (int i = 0; i < 8; ++i) cp[(long)i * N] = acc.f[i];
}

// ---------------------------------------------------------------------------
// Per-row scalar reductions: sx2=|X_r|^2, sy2=|Y_r|^2, sxy=X_r.Y_r for rows of
// two [nrows, 128] matrices.  One wave32 per row, float4 lane loads + shfl.
// ---------------------------------------------------------------------------
__global__ void row_scalars(const float* __restrict__ X,
                            const float* __restrict__ Y,
                            float* __restrict__ sx2, float* __restrict__ sy2,
                            float* __restrict__ sxy, int nrows) {
    int wave = blockIdx.x * (blockDim.x >> 5) + (threadIdx.x >> 5);
    int lane = threadIdx.x & 31;
    if (wave >= nrows) return;
    float4 x4 = ((const float4*)(X + (long)wave * Dd))[lane];
    float4 y4 = ((const float4*)(Y + (long)wave * Dd))[lane];
    float x2 = x4.x*x4.x + x4.y*x4.y + x4.z*x4.z + x4.w*x4.w;
    float y2 = y4.x*y4.x + y4.y*y4.y + y4.z*y4.z + y4.w*y4.w;
    float xy = x4.x*y4.x + x4.y*y4.y + x4.z*y4.z + x4.w*y4.w;
#pragma unroll
    for (int off = 16; off; off >>= 1) {
        x2 += __shfl_down(x2, off, 32);
        y2 += __shfl_down(y2, off, 32);
        xy += __shfl_down(xy, off, 32);
    }
    if (lane == 0) { sx2[wave] = x2; sy2[wave] = y2; sxy[wave] = xy; }
}

// ---------------------------------------------------------------------------
// Sentence projections: sm = sent_mean@W3+b3, sv = sent_div@W4+b4.  [B,1,D]
// ---------------------------------------------------------------------------
__global__ void sent_proj(const float* __restrict__ smean,
                          const float* __restrict__ sdiv,
                          const float* __restrict__ W3, const float* __restrict__ b3,
                          const float* __restrict__ W4, const float* __restrict__ b4,
                          float* __restrict__ sm, float* __restrict__ sv) {
    int b = blockIdx.x;
    int d = threadIdx.x;           // 128 threads
    float am = b3[d], av = b4[d];
    for (int h = 0; h < Hh; ++h) {
        am = fmaf(smean[b * Hh + h], W3[h * Dd + d], am);
        av = fmaf(sdiv [b * Hh + h], W4[h * Dd + d], av);
    }
    sm[b * Dd + d] = am;
    sv[b * Dd + d] = av;
}

// ---------------------------------------------------------------------------
// Rank-1 Gaussian JSD + softmax over S (pointwise; scalars precomputed).
// wsc* indexed by n=b*S+s (words: mq2,vq2,vqmq); lsc* by l (labels: p2,vp2,vpmp).
// ---------------------------------------------------------------------------
__global__ void jsd_rank1_softmax(const float* __restrict__ Gdot,
                                  const float* __restrict__ Ga2,
                                  const float* __restrict__ Ga3,
                                  const float* __restrict__ Ga4,
                                  const float* __restrict__ wsc_m2,
                                  const float* __restrict__ wsc_v2,
                                  const float* __restrict__ wsc_vm,
                                  const float* __restrict__ lsc_m2,
                                  const float* __restrict__ lsc_v2,
                                  const float* __restrict__ lsc_vm,
                                  float* __restrict__ score) {
    __shared__ float red[Ss];
    int bl = blockIdx.x;
    int b = bl >> 6, l = bl & 63;
    int s = threadIdx.x;           // 256 threads
    int n = b * Ss + s;
    float mq2 = wsc_m2[n], vq2 = wsc_v2[n], vqmq = wsc_vm[n];
    float p2  = lsc_m2[l], vp2 = lsc_v2[l], vpmp = lsc_vm[l];
    float nq = 1.f + vq2, np = 1.f + vp2;
    float dot = Gdot[(long)l * BS + n];
    float a2  = Ga2 [(long)l * BS + n];
    float a3  = Ga3 [(long)l * BS + n];
    float a4  = Ga4 [(long)l * BS + n];
    float d2    = mq2 - 2.f * a2 + p2;
    float vqd   = vqmq - a3;
    float vpd   = a4 - vpmp;
    float mahaq = d2 - vqd * vqd / nq;
    float mahap = d2 - vpd * vpd / np;
    const float DD = (float)Dd;
    float tr_pq = DD + (np - 1.f) - (nq - 1.f + dot * dot) / nq;
    float tr_qp = DD + (nq - 1.f) - (np - 1.f + dot * dot) / np;
    float kl_pq = 0.5f * (logf(nq) - logf(np) - DD + tr_pq + mahaq);
    float kl_qp = 0.5f * (logf(np) - logf(nq) - DD + tr_qp + mahap);
    float j = 0.5f * (kl_pq + kl_qp);
    red[s] = j; __syncthreads();
    for (int off = 128; off; off >>= 1) {
        if (s < off) red[s] = fmaxf(red[s], red[s + off]);
        __syncthreads();
    }
    float mx = red[0]; __syncthreads();
    float e = __expf(j - mx);
    red[s] = e; __syncthreads();
    for (int off = 128; off; off >>= 1) {
        if (s < off) red[s] += red[s + off];
        __syncthreads();
    }
    score[(long)bl * Ss + s] = e / red[0];
}

// ---------------------------------------------------------------------------
// text_div (St) builder: St[b,l] = I + 0.5*(sv svT + G^T G), G = sqrt(score)*wv.
// Both operand tiles staged through LDS with coalesced float4 row loads.
// ---------------------------------------------------------------------------
__global__ void st_build(const float* __restrict__ score,
                         const float* __restrict__ wv,
                         const float* __restrict__ sv,
                         float* __restrict__ St) {
    __shared__ float4 t4[32 * 4 * 2];        // tA | tB, 32x16 each
    __shared__ float ssc[32];
    float* tA = (float*)t4;
    float* tB = tA + 512;
    int bl = blockIdx.x;
    int b = bl >> 6;
    int tm = blockIdx.y, tn = blockIdx.z;
    int lane = threadIdx.x & 31;
    int m = lane & 15, g = lane >> 4;
    const float* sc  = score + (long)bl * Ss;
    const float* wvb = wv + (long)b * Ss * Dd;
    VC acc; acc.v = v8f{0.f,0.f,0.f,0.f,0.f,0.f,0.f,0.f};
    for (int kt = 0; kt < Ss; kt += 32) {
        int row = kt + lane;
        const float4* pa = (const float4*)(wvb + (long)row * Dd + tm * 16);
        const float4* pb = (const float4*)(wvb + (long)row * Dd + tn * 16);
#pragma unroll
        for (int q = 0; q < 4; ++q) {
            ((float4*)tA)[lane * 4 + q] = pa[q];
            ((float4*)tB)[lane * 4 + q] = pb[q];
        }
        ssc[lane] = sqrtf(sc[row]);
        __syncthreads();
        VA ua, ub;
#pragma unroll
        for (int h = 0; h < 16; ++h) {
            int ka = b_kidx(h, g);                       // 0..31 (local)
            ua.h[h] = (_Float16)(ssc[ka] * tA[ka * 16 + m]);
            int kb = b_kidx(h, g);
            ub.h[h] = (_Float16)(ssc[kb] * tB[kb * 16 + m]);
        }
        acc.v = wmma_f16(ua.v, ub.v, acc.v);
        __syncthreads();
    }
    const float* svb = sv + b * Dd;
    int c = tn * 16 + (lane & 15);
    float svc = svb[c];
    float* cp = St + (long)bl * Dd * Dd + (long)(tm * 16 + 8 * g) * Dd + c;
#pragma unroll
    for (int i = 0; i < 8; ++i) {
        int r = tm * 16 + i + 8 * g;
        cp[(long)i * Dd] = ((r == c) ? 1.f : 0.f) + 0.5f * (svb[r] * svc + acc.f[i]);
    }
}

// ---------------------------------------------------------------------------
// Full-covariance symmetric JSD vs rank-1 label covariance (one block / (b,l)).
// St pulled into LDS via gfx1250 async DMA (ASYNCcnt), then Cholesky in LDS.
// ---------------------------------------------------------------------------
__device__ __forceinline__ float block_reduce128(float v, float* sred, int t) {
    sred[t] = v; __syncthreads();
    for (int off = 64; off; off >>= 1) {
        if (t < off) sred[t] += sred[t + off];
        __syncthreads();
    }
    float r = sred[0]; __syncthreads();
    return r;
}

__global__ void __launch_bounds__(128)
jsd_full_kernel(const float* __restrict__ St,
                const float* __restrict__ vmean,
                const float* __restrict__ sm,
                const float* __restrict__ label_mean,
                const float* __restrict__ label_div,
                float* __restrict__ relate) {
    extern __shared__ float sA[];          // 128*128 floats (64 KB)
    __shared__ float sred[Dd];
    __shared__ float sU[Dd], sDelta[Dd], sZ[Dd], sY[Dd];
    int bl = blockIdx.x;
    int b = bl >> 6, l = bl & 63;
    int t = threadIdx.x;                   // 128 threads

    const float* A0 = St + (long)bl * Dd * Dd;
    // async global -> LDS copy, 16B per lane per issue (ASYNCcnt-tracked)
    for (int i = t * 4; i < Dd * Dd; i += Dd * 4) {
        unsigned lo = (unsigned)(unsigned long long)(&sA[i]);
        const float* ga = A0 + i;
        asm volatile("global_load_async_to_lds_b128 %0, %1, off"
                     :: "v"(lo), "v"(ga) : "memory");
    }
    asm volatile("s_wait_asynccnt 0x0" ::: "memory");
    float u = label_div[l * Dd + t];
    sU[t] = u;
    float tmean = 0.5f * (sm[b * Dd + t] + vmean[(long)bl * Dd + t]);
    float del = label_mean[l * Dd + t] - tmean;
    sDelta[t] = del;
    __syncthreads();

    // Scalars needing the original (pre-Cholesky) matrix:
    float colsum = 0.f;
    for (int i = 0; i < Dd; ++i) colsum = fmaf(sU[i], sA[i * Dd + t], colsum);
    float trSp   = block_reduce128(sA[t * Dd + t], sred, t);
    float uSpu   = block_reduce128(sU[t] * colsum, sred, t);
    float u2     = block_reduce128(u * u, sred, t);
    float delta2 = block_reduce128(del * del, sred, t);
    float udelta = block_reduce128(u * del, sred, t);

    // In-place Cholesky (lower), column-parallel right-looking.
    for (int k = 0; k < Dd; ++k) {
        if (t == 0) sA[k * Dd + k] = sqrtf(sA[k * Dd + k]);
        __syncthreads();
        float lkk = sA[k * Dd + k];
        if (t > k) sA[t * Dd + k] /= lkk;
        __syncthreads();
        if (t > k) {
            float ajk = sA[t * Dd + k];
            for (int i = t; i < Dd; ++i)
                sA[i * Dd + t] -= sA[i * Dd + k] * ajk;
        }
        __syncthreads();
    }
    float logdet_p = block_reduce128(2.f * logf(sA[t * Dd + t]), sred, t);

    // Forward solves L z = u and L y = delta (serialized inside one wave).
    if (t == 0) {
        for (int i = 0; i < Dd; ++i) {
            float s = sU[i];
            for (int j = 0; j < i; ++j) s -= sA[i * Dd + j] * sZ[j];
            sZ[i] = s / sA[i * Dd + i];
        }
    }
    if (t == 1) {
        for (int i = 0; i < Dd; ++i) {
            float s = sDelta[i];
            for (int j = 0; j < i; ++j) s -= sA[i * Dd + j] * sY[j];
            sY[i] = s / sA[i * Dd + i];
        }
    }
    __syncthreads();

    // tr(Sp^-1) = ||L^-1||_F^2 : thread t builds column t of L^-1 (scratch).
    float w[Dd];
    float s2 = 0.f;
    for (int i = t; i < Dd; ++i) {
        float s = (i == t) ? 1.f : 0.f;
        for (int k = t; k < i; ++k) s -= sA[i * Dd + k] * w[k];
        float wi = s / sA[i * Dd + i];
        w[i] = wi;
        s2 = fmaf(wi, wi, s2);
    }
    float trSpInv = block_reduce128(s2, sred, t);
    float uSpInvu = block_reduce128(sZ[t] * sZ[t], sred, t);   // ||L^-1 u||^2
    float mahap   = block_reduce128(sY[t] * sY[t], sred, t);   // ||L^-1 d||^2

    if (t == 0) {
        const float DD = (float)Dd;
        float np = 1.f + u2;
        float logdet_q = logf(np);
        float tr_qinv_p = trSp - uSpu / np;                 // tr(Sq^-1 Sp)
        float mahaq     = delta2 - udelta * udelta / np;    // d^T Sq^-1 d
        float tr_pinv_q = trSpInv + uSpInvu;                // tr(Sp^-1 Sq)
        float kl_pq = 0.5f * (logdet_q - logdet_p - DD + tr_qinv_p + mahaq);
        float kl_qp = 0.5f * (logdet_p - logdet_q - DD + tr_pinv_q + mahap);
        relate[bl] = 0.5f * (kl_pq + kl_qp);
    }
}

// ---------------------------------------------------------------------------
// Broadcast outputs, non-temporal float4 stores:
//   lm_b [B,L,D] = label_mean;  lc_b [B,L,D,D] = I + u u^T.
// ---------------------------------------------------------------------------
__global__ void fill_outputs(const float* __restrict__ label_mean,
                             const float* __restrict__ label_div,
                             float* __restrict__ out) {
    long stride = (long)gridDim.x * blockDim.x;
    long idx0 = (long)blockIdx.x * blockDim.x + threadIdx.x;
    const long n1v = (long)Bb * Ll * Dd / 4;             // 16384 float4
    f4* o1 = (f4*)(out + Bb * Ll);
    const f4* lmv = (const f4*)label_mean;
    for (long i = idx0; i < n1v; i += stride)
        __builtin_nontemporal_store(lmv[i & (Ll * Dd / 4 - 1)], &o1[i]);
    const long n2v = (long)Bb * Ll * Dd * Dd / 4;        // 2097152 float4
    f4* o2 = o1 + n1v;
    for (long i = idx0; i < n2v; i += stride) {
        long e = i << 2;
        int jj = (int)(e & 127);
        int ii = (int)((e >> 7) & 127);
        int l  = (int)((e >> 14) & 63);
        const float* ur = label_div + l * Dd;
        float ui = ur[ii];
        f4 uj = *(const f4*)(ur + jj);
        f4 r = ui * uj;
        if (ii >= jj && ii < jj + 4) {
            if (ii == jj)          r.x += 1.f;
            else if (ii == jj + 1) r.y += 1.f;
            else if (ii == jj + 2) r.z += 1.f;
            else                   r.w += 1.f;
        }
        __builtin_nontemporal_store(r, &o2[i]);
    }
}

// ---------------------------------------------------------------------------
extern "C" void kernel_launch(void* const* d_in, const int* in_sizes, int n_in,
                              void* d_out, int out_size, void* d_ws, size_t ws_size,
                              hipStream_t stream) {
    const float* word_mean = (const float*)d_in[0];
    const float* word_div  = (const float*)d_in[1];
    const float* sent_mean = (const float*)d_in[2];
    const float* sent_div  = (const float*)d_in[3];
    const float* W1 = (const float*)d_in[4];
    const float* b1 = (const float*)d_in[5];
    const float* W2 = (const float*)d_in[6];
    const float* b2 = (const float*)d_in[7];
    const float* W3 = (const float*)d_in[8];
    const float* b3 = (const float*)d_in[9];
    const float* W4 = (const float*)d_in[10];
    const float* b4 = (const float*)d_in[11];
    const float* label_mean = (const float*)d_in[12];
    const float* label_div  = (const float*)d_in[13];
    float* out = (float*)d_out;
    float* ws  = (float*)d_ws;

    // workspace layout (floats)
    float* wm    = ws;                       // 262144
    float* wv    = wm + (long)BS * Dd;       // 262144
    float* sm    = wv + (long)BS * Dd;       // 1024
    float* sv    = sm + Bb * Dd;             // 1024
    float* Gdot  = sv + Bb * Dd;             // 131072 each
    float* Ga2   = Gdot + (long)Ll * BS;
    float* Ga3   = Ga2  + (long)Ll * BS;
    float* Ga4   = Ga3  + (long)Ll * BS;
    float* score = Ga4  + (long)Ll * BS;     // 131072
    float* vmean = score + (long)BL * Ss;    // 65536
    float* Stw   = vmean + (long)BL * Dd;    // 8388608
    float* wsc0  = Stw + (long)BL * Dd * Dd; // word scalars: 2048 each
    float* wsc1  = wsc0 + BS;
    float* wsc2  = wsc1 + BS;
    float* lsc0  = wsc2 + BS;                // label scalars: 64 each
    float* lsc1  = lsc0 + Ll;
    float* lsc2  = lsc1 + Ll;

    // 1) word projections: [BS,H] @ [H,D] + bias  (WMMA)
    wmma_gemm_nn<<<dim3(BS / 16, Dd / 16, 1), 32, 0, stream>>>(
        word_mean, W1, b1, wm, BS, Dd, Hh, 0, 0, 0);
    wmma_gemm_nn<<<dim3(BS / 16, Dd / 16, 1), 32, 0, stream>>>(
        word_div, W2, b2, wv, BS, Dd, Hh, 0, 0, 0);

    // 2) sentence projections + per-row scalars
    sent_proj<<<Bb, Dd, 0, stream>>>(sent_mean, sent_div, W3, b3, W4, b4, sm, sv);
    row_scalars<<<BS / 8, 256, 0, stream>>>(wm, wv, wsc0, wsc1, wsc2, BS);
    row_scalars<<<Ll / 8, 256, 0, stream>>>(label_mean, label_div,
                                            lsc0, lsc1, lsc2, Ll);

    // 3) four label-vs-word contractions 'ld,bsd->bls' (WMMA NT)
    wmma_gemm_nt<<<dim3(Ll / 16, BS / 16, 1), 32, 0, stream>>>(label_div,  wv, Gdot, Ll, BS, Dd);
    wmma_gemm_nt<<<dim3(Ll / 16, BS / 16, 1), 32, 0, stream>>>(label_mean, wm, Ga2,  Ll, BS, Dd);
    wmma_gemm_nt<<<dim3(Ll / 16, BS / 16, 1), 32, 0, stream>>>(label_mean, wv, Ga3,  Ll, BS, Dd);
    wmma_gemm_nt<<<dim3(Ll / 16, BS / 16, 1), 32, 0, stream>>>(label_div,  wm, Ga4,  Ll, BS, Dd);

    // 4) rank-1 JSD + softmax over S (pointwise)
    jsd_rank1_softmax<<<BL, Ss, 0, stream>>>(Gdot, Ga2, Ga3, Ga4,
                                             wsc0, wsc1, wsc2,
                                             lsc0, lsc1, lsc2, score);

    // 5) v_mean = score @ wm per batch (WMMA, batched)
    wmma_gemm_nn<<<dim3(Ll / 16, Dd / 16, Bb), 32, 0, stream>>>(
        score, wm, nullptr, vmean, Ll, Dd, Ss,
        (long)Ll * Ss, (long)Ss * Dd, (long)Ll * Dd);

    // 6) text_div = I + 0.5*(sv sv^T + G^T G) per (b,l) (WMMA)
    st_build<<<dim3(BL, Dd / 16, Dd / 16), 32, 0, stream>>>(score, wv, sv, Stw);

    // 7) full-covariance symmetric JSD (async-LDS + Cholesky) -> relate_score
    jsd_full_kernel<<<BL, Dd, Dd * Dd * sizeof(float), stream>>>(
        Stw, vmean, sm, label_mean, label_div, out);

    // 8) broadcast lm_b / lc_b outputs (non-temporal stores)
    fill_outputs<<<2048, 256, 0, stream>>>(label_mean, label_div, out);
}
